// BlockSparseAttention_90640989815089
// MI455X (gfx1250) — matrix operand order repaired and checked by reading, not compiled, not run on hardware
//
#include <hip/hip_runtime.h>

// ---------------------------------------------------------------------------
// Block-sparse (tridiagonal-block) attention for MI455X (gfx1250).
//
// Roofline: attn materialization (537 MB f32 NT stores) dominates: ~23 us
// floor at 23.3 TB/s HBM vs ~3.2 GFLOP of band-limited matmul. We use the
// codegen-confirmed v_wmma_f32_16x16x32_f16 (f16 in, f32 accum), restrict all
// data movement to the 192-wide key band, stream every large output with
// non-temporal hints (TH=NT) so the 192 MB L2 keeps the K/V band for reuse by
// adjacent q-blocks, and zero-fill the out-of-band attn region with a
// 128-bit NT grid-stride kernel.
// ---------------------------------------------------------------------------

typedef __attribute__((ext_vector_type(16))) _Float16 v16h;
typedef __attribute__((ext_vector_type(8)))  float    v8f;
typedef __attribute__((ext_vector_type(4)))  float    v4f;   // native vec for NT stores

#define NUM_H 16
#define SEQ_T 2048
#define HD    64
#define QBLK  64
#define BANDW 192   // max key-band width: 3 blocks of 64

// ---- WMMA operand loaders -------------------------------------------------
// A-matrix 16x32 f16 layout (ISA 7.12.2): lane L<16 -> row M=L, halves
// {0..7} = K 0..7, {8..15} = K 16..23; lanes 16..31 -> same rows, K 8..15 /
// 24..31.  So per lane: two 8-element contiguous runs at col0+kb, col0+kb+16.
__device__ __forceinline__ v16h load_a_f32(const float* __restrict__ base,
                                           int ld, int row0, int col0) {
  const int lane = threadIdx.x & 31;
  const int row  = row0 + (lane & 15);
  const int kb   = (lane >> 4) << 3;          // 0 or 8
  const float* p = base + row * ld + col0 + kb;
  v16h a;
#pragma unroll
  for (int j = 0; j < 8; ++j) {
    a[j]     = (_Float16)p[j];
    a[j + 8] = (_Float16)p[j + 16];
  }
  return a;
}

__device__ __forceinline__ v16h load_a_lds(const _Float16* __restrict__ base,
                                           int ld, int row0, int col0) {
  const int lane = threadIdx.x & 31;
  const int row  = row0 + (lane & 15);
  const int kb   = (lane >> 4) << 3;
  const _Float16* p = base + row * ld + col0 + kb;
  v16h a;
#pragma unroll
  for (int j = 0; j < 8; ++j) {
    a[j]     = p[j];
    a[j + 8] = p[j + 16];
  }
  return a;
}

// B-matrix 32x16 f16 layout (matches A): lanes 0-15 hold K 0..7, 16..23 for
// col N=lane; lanes 16-31 hold K 8..15, 24..31.  Memory is stored row-major
// by N (K-dim contiguous), so two 8-element runs at k0+kh, k0+kh+16.
__device__ __forceinline__ v16h load_b_f32(const float* __restrict__ base,
                                           int ld, int n0, int k0) {
  const int lane = threadIdx.x & 31;
  const int row  = n0 + (lane & 15);
  const int kh   = (lane >> 4) << 3;          // 0 or 8
  const float* p = base + row * ld + k0 + kh;
  v16h b;
#pragma unroll
  for (int j = 0; j < 16; ++j) b[j] = (_Float16)p[j < 8 ? j : j + 8];
  return b;
}

__device__ __forceinline__ v16h load_b_lds(const _Float16* __restrict__ base,
                                           int ld, int n0, int k0) {
  const int lane = threadIdx.x & 31;
  const int row  = n0 + (lane & 15);
  const int kh   = (lane >> 4) << 3;
  const _Float16* p = base + row * ld + k0 + kh;
  v16h b;
#pragma unroll
  for (int j = 0; j < 16; ++j) b[j] = p[j < 8 ? j : j + 8];
  return b;
}

// ---- out-of-band attn zero-fill (pure streaming-store kernel) -------------
__global__ void BSA_zero_attn(v4f* __restrict__ p, size_t n4) {
  size_t i      = (size_t)blockIdx.x * blockDim.x + threadIdx.x;
  size_t stride = (size_t)gridDim.x * blockDim.x;
  const v4f z = {0.f, 0.f, 0.f, 0.f};
  for (; i < n4; i += stride)
    __builtin_nontemporal_store(z, p + i);     // TH=NT b128: don't pollute L2
}

// ---- main attention kernel: one block per (b, h, 64-row q-block) ----------
__global__ __launch_bounds__(128)
void BSA_band_attn(const float* __restrict__ q, const float* __restrict__ k,
                   const float* __restrict__ v,
                   const unsigned char* __restrict__ mask,
                   float* __restrict__ out, float* __restrict__ attn) {
  __shared__ _Float16 Vt[HD * BANDW];    // 24 KB: V band transposed [d][k]
  __shared__ _Float16 S [QBLK * BANDW];  // 24 KB: f16 scores, then P in-place

  const int tid  = threadIdx.x;
  const int lane = tid & 31;
  const int wave = tid >> 5;

  const int qb = blockIdx.x & 31;                 // T/QBLK == 32
  const int h  = (blockIdx.x >> 5) & (NUM_H - 1);
  const int b  = blockIdx.x >> 9;                 // / (32*16)

  const int kb0 = (qb > 0  ? qb - 1 : 0)  * QBLK; // band start
  const int kb1 = (qb < 31 ? qb + 2 : 32) * QBLK; // band end
  const int nk  = kb1 - kb0;                      // 128 or 192

  const size_t bh = (size_t)(b * NUM_H + h);
  const float* qp = q + (bh * SEQ_T + (size_t)qb * QBLK) * HD;
  const float* kp = k + (bh * SEQ_T + (size_t)kb0) * HD;
  const float* vp = v + (bh * SEQ_T + (size_t)kb0) * HD;
  float* op = out  + (bh * SEQ_T + (size_t)qb * QBLK) * HD;
  float* ap = attn + (bh * SEQ_T + (size_t)qb * QBLK) * (size_t)SEQ_T;

  // Stage V band transposed into LDS (f32 -> f16) for contiguous B reads.
  for (int i = tid; i < nk * HD; i += 128) {
    const int kk = i >> 6;            // / HD
    const int d  = i & (HD - 1);
    Vt[d * BANDW + kk] = (_Float16)vp[kk * HD + d];
  }
  __syncthreads();

  const int q0 = wave * 16;           // this wave's 16 query rows
  const int n  = lane & 15;           // C/D layout: column = lane%16
  const int mh = (lane >> 4) << 3;    // C/D layout: row block 0 or 8

  // ---- scores = scale * Q K^T over the band, masked, to LDS (f16) --------
  const int nkt = nk >> 4;
  for (int kt = 0; kt < nkt; ++kt) {
    v8f acc = {};
#pragma unroll
    for (int dc = 0; dc < HD; dc += 32) {
      v16h A = load_a_f32(qp, HD, q0, dc);       // Q[q][d]
      v16h B = load_b_f32(kp, HD, kt * 16, dc);  // K[key][d] (key = N)
      acc = __builtin_amdgcn_wmma_f32_16x16x32_f16(
          false, A, false, B, (short)0, acc, false, false);
    }
    const int   kcol = kt * 16 + n;
    // external key mask (B,1,1,T): masked -> large negative (exp underflows)
    const float mval =
        mask[(size_t)b * SEQ_T + kb0 + kcol] ? 0.0f : -30000.0f;
#pragma unroll
    for (int r = 0; r < 8; ++r)
      S[(q0 + mh + r) * BANDW + kcol] = (_Float16)(acc[r] * 0.125f + mval);
  }
  __syncthreads();

  // ---- per-row softmax: a lane *pair* per row (all 128 threads active) ---
  // Threads 2r and 2r+1 are adjacent lanes of the same wave32; each handles
  // half the band and partials combine with shfl_xor(1).
  {
    const int row  = tid >> 1;                  // 0..63
    const int half = tid & 1;
    const int hw   = nk >> 1;                   // 64 or 96 columns per thread
    const int c0   = half * hw;
    const int c1   = c0 + hw;
    _Float16* srow = S + row * BANDW;

    float m = -1e30f;
    for (int c = c0; c < c1; ++c) m = fmaxf(m, (float)srow[c]);
    m = fmaxf(m, __shfl_xor(m, 1, 32));

    float sum = 0.0f;
    for (int c = c0; c < c1; ++c) sum += __expf((float)srow[c] - m);
    sum += __shfl_xor(sum, 1, 32);
    const float inv = 1.0f / sum;

    float* arow = ap + (size_t)row * SEQ_T + kb0;
    for (int c = c0; c < c1; ++c) {
      const float p = __expf((float)srow[c] - m) * inv;
      __builtin_nontemporal_store(p, arow + c);  // streamed attn band
      srow[c] = (_Float16)p;   // in-place P: same element, same owner thread
    }
  }
  __syncthreads();

  // ---- output = P V -------------------------------------------------------
  for (int nt = 0; nt < 4; ++nt) {   // 4 x 16 output columns = HD
    v8f acc = {};
    for (int kc = 0; kc < nk; kc += 32) {
      v16h A = load_a_lds(S,  BANDW, q0,      kc);  // P[q][k]
      v16h B = load_b_lds(Vt, BANDW, nt * 16, kc);  // Vt[d][k] (d = N)
      acc = __builtin_amdgcn_wmma_f32_16x16x32_f16(
          false, A, false, B, (short)0, acc, false, false);
    }
#pragma unroll
    for (int r = 0; r < 8; ++r)
      __builtin_nontemporal_store(
          acc[r], &op[(q0 + mh + r) * HD + nt * 16 + n]);
  }
}

// ---------------------------------------------------------------------------
extern "C" void kernel_launch(void* const* d_in, const int* in_sizes, int n_in,
                              void* d_out, int out_size, void* d_ws,
                              size_t ws_size, hipStream_t stream) {
  const float* q = (const float*)d_in[0];
  const float* k = (const float*)d_in[1];
  const float* v = (const float*)d_in[2];
  const unsigned char* mask = (const unsigned char*)d_in[3];  // jax bool = u8

  const int B = in_sizes[0] / (NUM_H * SEQ_T * HD);           // == 2

  float* out = (float*)d_out;
  const size_t out_elems  = (size_t)B * NUM_H * SEQ_T * HD;
  float*       attn       = out + out_elems;
  const size_t attn_elems = (size_t)B * NUM_H * SEQ_T * SEQ_T;

  // 1) zero the dense attn tensor (out-of-band entries are exactly 0)
  BSA_zero_attn<<<8192, 256, 0, stream>>>((v4f*)attn, attn_elems / 4);

  // 2) band attention: B*H*(T/64) workgroups of 4 waves
  dim3 grid((unsigned)(B * NUM_H * (SEQ_T / QBLK)));
  BSA_band_attn<<<grid, 128, 0, stream>>>(q, k, v, mask, out, attn);
}
